// DMPNNEncoder_1855425872153
// MI455X (gfx1250) — matrix operand order repaired and checked by reading, not compiled, not run on hardware
//
#include <hip/hip_runtime.h>

// ---------------------------------------------------------------------------
// D-MPNN encoder for MI455X (gfx1250, wave32, WMMA).
// h stored bf16 [E,128]; all GEMMs via v_wmma_f32_16x16x32_bf16 (fp32 acc).
// Blocks process 64 edges: B fragments live in registers across 4 M-tiles
// (4x fewer B loads per wmma). Segment sums in fp32 via global atomics
// (sum_in is 25.6MB -> L2 resident on the 192MB L2).
// ---------------------------------------------------------------------------

typedef __bf16 bf16_t;
typedef __bf16        v16bf  __attribute__((ext_vector_type(16)));
typedef float         v8f    __attribute__((ext_vector_type(8)));
typedef unsigned int  u32x4  __attribute__((ext_vector_type(4)));

#define H_DIM   128
#define FA_DIM  133
#define FB_DIM  14
#define K_INIT  147          // real K for init GEMM (padded to 160)
#define DEPTH_I 3

// ---- fragment helpers -----------------------------------------------------
// 16-bit A fragment (16x32, MxK), per ISA 7.12.2:
//  lanes 0-15: M=lane, v0..3 hold K=0..7, v4..7 hold K=16..23
//  lanes16-31: M=lane-16, v0..3 hold K=8..15, v4..7 hold K=24..31
// => per lane: two contiguous 16B chunks at byte offsets kt*64 + (hi?16:0), +32.
static __device__ __forceinline__ v16bf load_a_frag(const bf16_t* ldsRow0,
                                                    int rowStrideElems,
                                                    int ktile, int lane) {
  int row = lane & 15;
  const char* p = (const char*)(ldsRow0 + row * rowStrideElems)
                  + ktile * 64 + ((lane & 16) ? 16 : 0);
  union { u32x4 q[2]; v16bf v; } f;
  f.q[0] = *(const u32x4*)p;
  f.q[1] = *(const u32x4*)(p + 32);
  return f.v;
}

// 16-bit B fragment (32x16, KxN): lane n=lane&15, same K order as A:
// lanes0-15 K=0..7,16..23, lanes16-31 K=8..15,24..31. Pre-packed so each
// lane's 8 dwords are contiguous (two b128 loads).
static __device__ __forceinline__ v16bf load_b_frag(const unsigned* pw,
                                                    int kt, int nt, int NT,
                                                    int lane) {
  const u32x4* p = (const u32x4*)(pw + (((kt * NT + nt) << 5) + lane) * 8);
  union { u32x4 q[2]; v16bf v; } f;
  f.q[0] = p[0];
  f.q[1] = p[1];
  return f.v;
}

static __device__ __forceinline__ float bfbits2f(unsigned short b) {
  union { unsigned u; float f; } c; c.u = ((unsigned)b) << 16; return c.f;
}

// ---- weight packing: fp32 [H,K] row-major -> bf16 B-fragment layout --------
__global__ __launch_bounds__(256) void k_pack_w(const float* __restrict__ W,
                                                unsigned* __restrict__ pw,
                                                int Kreal, int KT, int NT) {
  int p = blockIdx.x * 256 + threadIdx.x;
  int total = KT * NT * 256;
  if (p >= total) return;
  int v    = p & 7;
  int lane = (p >> 3) & 31;
  int nt   = (p >> 8) % NT;
  int kt   = p / (NT << 8);
  int n    = nt * 16 + (lane & 15);
  int k0   = kt * 32 + ((lane & 16) ? 8 : 0) + 2 * v + ((v >= 4) ? 8 : 0);
  float f0 = (k0     < Kreal) ? W[(size_t)n * Kreal + k0]     : 0.0f;
  float f1 = (k0 + 1 < Kreal) ? W[(size_t)n * Kreal + k0 + 1] : 0.0f;
  union { bf16_t b[2]; unsigned u; } c;
  c.b[0] = (bf16_t)f0;
  c.b[1] = (bf16_t)f1;
  pw[p] = c.u;
}

// ---- edge init: h0 = relu(W_init @ concat(atom[src], bond) + b) ------------
// Block = 64 edges, 8 waves (one 16-ch n-tile each), 4 M-tiles per wave.
__global__ __launch_bounds__(256) void k_edge_init(
    const float* __restrict__ atom, const float* __restrict__ bond,
    const int* __restrict__ srcIdx,
    const unsigned* __restrict__ pwInit, const float* __restrict__ bInit,
    bf16_t* __restrict__ h0) {
  __shared__ __align__(16) bf16_t sA[64][176];   // 64 edges x (K padded 160, +pad)
  __shared__ int sSrc[64];
  const int e0 = blockIdx.x << 6;

  if (threadIdx.x < 64) sSrc[threadIdx.x] = srcIdx[e0 + threadIdx.x];
  __syncthreads();

  for (int i = threadIdx.x; i < 64 * 176; i += 256) {
    int r = i / 176, k = i % 176;
    float v = 0.0f;
    if (k < FA_DIM)       v = atom[(size_t)sSrc[r] * FA_DIM + k];
    else if (k < K_INIT)  v = bond[(size_t)(e0 + r) * FB_DIM + (k - FA_DIM)];
    sA[r][k] = (bf16_t)v;
  }
  __syncthreads();

  const int lane = threadIdx.x & 31;
  const int nt   = threadIdx.x >> 5;          // 8 waves -> 8 n-tiles = 128 ch
  const int col  = nt * 16 + (lane & 15);
  const float bias  = bInit[col];
  const int   rbase = (lane & 16) ? 8 : 0;

  v16bf bfr[5];                                // B resident across all M-tiles
#pragma unroll
  for (int kt = 0; kt < 5; ++kt) bfr[kt] = load_b_frag(pwInit, kt, nt, 8, lane);

#pragma unroll
  for (int mt = 0; mt < 4; ++mt) {
    v8f acc = {0, 0, 0, 0, 0, 0, 0, 0};
#pragma unroll
    for (int kt = 0; kt < 5; ++kt) {
      v16bf a = load_a_frag(&sA[mt * 16][0], 176, kt, lane);
      acc = __builtin_amdgcn_wmma_f32_16x16x32_bf16(false, a, false, bfr[kt],
                                                    (short)0, acc, false, false);
    }
#pragma unroll
    for (int r = 0; r < 8; ++r) {
      float f = acc[r] + bias;
      f = f > 0.0f ? f : 0.0f;
      h0[(size_t)(e0 + mt * 16 + rbase + r) * H_DIM + col] = (bf16_t)f;
    }
  }
}

// ---- segment sum: sum[dst[e]] += h[e]  (bf16 -> fp32 atomics) --------------
__global__ __launch_bounds__(256) void k_segsum(const bf16_t* __restrict__ h,
                                                const int* __restrict__ dstIdx,
                                                float* __restrict__ sum, int E) {
  long long t = (long long)blockIdx.x * 256 + threadIdx.x;
  if (t >= (long long)E * 64) return;
  int e    = (int)(t >> 6);
  int pair = (int)(t & 63);
  unsigned u = ((const unsigned*)h)[(size_t)e * 64 + pair];
  union { unsigned u; float f; } lo, hi;
  lo.u = u << 16;
  hi.u = u & 0xffff0000u;
  float* base = sum + (size_t)dstIdx[e] * H_DIM + pair * 2;
  atomicAdd(base,     lo.f);
  atomicAdd(base + 1, hi.f);
}

// ---- update: h' = relu(W_upd @ [h | sum_in[src] - h[e^1]] + b) -------------
// Block = 64 edges; B (8 k-tiles) resident in registers for 4 M-tiles.
__global__ __launch_bounds__(256) void k_update(
    const bf16_t* __restrict__ hin, bf16_t* __restrict__ hout,
    const float* __restrict__ sumIn, const int* __restrict__ srcIdx,
    const unsigned* __restrict__ pwUpd, const float* __restrict__ bUpd) {
  __shared__ __align__(16) bf16_t sA[64][264];   // 64 edges x (K=256, +pad)
  __shared__ int sSrc[64];
  const int e0 = blockIdx.x << 6;

  if (threadIdx.x < 64) sSrc[threadIdx.x] = srcIdx[e0 + threadIdx.x];
  __syncthreads();

  // K 0..127 : copy h rows (already bf16) as dwords
  const unsigned* h32 = (const unsigned*)hin;
  for (int i = threadIdx.x; i < 64 * 64; i += 256) {
    int r = i >> 6, c = i & 63;
    ((unsigned*)&sA[r][0])[c] = h32[(size_t)(e0 + r) * 64 + c];
  }
  // K 128..255 : m = sum_in[src[e]] - h[e^1]   (rev pair stays in this tile)
  const unsigned short* h16 = (const unsigned short*)hin;
  for (int i = threadIdx.x; i < 64 * 128; i += 256) {
    int r = i >> 7, k = i & 127;
    float sv = sumIn[(size_t)sSrc[r] * H_DIM + k];
    float hr = bfbits2f(h16[(size_t)((e0 + r) ^ 1) * H_DIM + k]);
    sA[r][128 + k] = (bf16_t)(sv - hr);
  }
  __syncthreads();

  const int lane = threadIdx.x & 31;
  const int nt   = threadIdx.x >> 5;
  const int col  = nt * 16 + (lane & 15);
  const float bias  = bUpd[col];
  const int   rbase = (lane & 16) ? 8 : 0;

  v16bf bfr[8];                                // 64 VGPRs of B, loaded once
#pragma unroll
  for (int kt = 0; kt < 8; ++kt) bfr[kt] = load_b_frag(pwUpd, kt, nt, 8, lane);

#pragma unroll
  for (int mt = 0; mt < 4; ++mt) {
    v8f acc = {0, 0, 0, 0, 0, 0, 0, 0};
#pragma unroll
    for (int kt = 0; kt < 8; ++kt) {
      v16bf a = load_a_frag(&sA[mt * 16][0], 264, kt, lane);
      acc = __builtin_amdgcn_wmma_f32_16x16x32_bf16(false, a, false, bfr[kt],
                                                    (short)0, acc, false, false);
    }
#pragma unroll
    for (int r = 0; r < 8; ++r) {
      float f = acc[r] + bias;
      f = f > 0.0f ? f : 0.0f;
      hout[(size_t)(e0 + mt * 16 + rbase + r) * H_DIM + col] = (bf16_t)f;
    }
  }
}

// ---- readout: out[j] = sum_n relu(W_read @ atom_sum[n] + b)[j] -------------
__global__ __launch_bounds__(256) void k_readout(
    const float* __restrict__ atomSum, const unsigned* __restrict__ pwRead,
    const float* __restrict__ bRead, float* __restrict__ out, int N) {
  __shared__ __align__(16) bf16_t sA[16][136];
  const int   lane  = threadIdx.x & 31;
  const int   nt    = threadIdx.x >> 5;
  const int   col   = nt * 16 + (lane & 15);
  const float bias  = bRead[col];
  const int   rbase = (lane & 16) ? 8 : 0;
  float csum = 0.0f;

  v16bf bfr[4];                                // B loaded once per wave
#pragma unroll
  for (int kt = 0; kt < 4; ++kt) bfr[kt] = load_b_frag(pwRead, kt, nt, 8, lane);

  const int ntiles = (N + 15) >> 4;
  for (int tile = blockIdx.x; tile < ntiles; tile += gridDim.x) {
    const int a0 = tile << 4;
    __syncthreads();
    for (int i = threadIdx.x; i < 16 * 128; i += 256) {
      int r = i >> 7, k = i & 127;
      int a = a0 + r;
      sA[r][k] = (bf16_t)((a < N) ? atomSum[(size_t)a * H_DIM + k] : 0.0f);
    }
    __syncthreads();
    v8f acc = {0, 0, 0, 0, 0, 0, 0, 0};
#pragma unroll
    for (int kt = 0; kt < 4; ++kt) {
      v16bf a = load_a_frag(&sA[0][0], 136, kt, lane);
      acc = __builtin_amdgcn_wmma_f32_16x16x32_bf16(false, a, false, bfr[kt],
                                                    (short)0, acc, false, false);
    }
#pragma unroll
    for (int r = 0; r < 8; ++r) {
      int a = a0 + rbase + r;
      if (a < N) {                      // mask pad rows (relu(bias) != 0)
        float f = acc[r] + bias;
        csum += (f > 0.0f ? f : 0.0f);
      }
    }
  }
  csum += __shfl_xor(csum, 16, 32);     // combine lane l and l+16 (same col)
  if (lane < 16) atomicAdd(&out[nt * 16 + lane], csum);
}

// ---------------------------------------------------------------------------
extern "C" void kernel_launch(void* const* d_in, const int* in_sizes, int n_in,
                              void* d_out, int out_size, void* d_ws, size_t ws_size,
                              hipStream_t stream) {
  (void)n_in; (void)ws_size;
  const float* atom  = (const float*)d_in[0];
  const float* bond  = (const float*)d_in[1];
  const int*   eidx  = (const int*)d_in[2];     // [2,E]: row0=src, row1=dst
  const float* Winit = (const float*)d_in[3];
  const float* binit = (const float*)d_in[4];
  const float* Wupd  = (const float*)d_in[5];
  const float* bupd  = (const float*)d_in[6];
  const float* Wread = (const float*)d_in[7];
  const float* bread = (const float*)d_in[8];

  const int N = in_sizes[0] / FA_DIM;
  const int E = in_sizes[1] / FB_DIM;
  const int* srcIdx = eidx;
  const int* dstIdx = eidx + E;

  // workspace carve-up
  char*  ws  = (char*)d_ws;
  size_t szH = (size_t)E * H_DIM * sizeof(bf16_t);   // 134 MB each
  size_t szS = (size_t)N * H_DIM * sizeof(float);    // 25.6 MB
  bf16_t*   hA     = (bf16_t*)ws;
  bf16_t*   hB     = (bf16_t*)(ws + szH);
  float*    sumIn  = (float*)(ws + 2 * szH);
  unsigned* pwInit = (unsigned*)(ws + 2 * szH + szS);   // 5*8*256 dwords
  unsigned* pwUpd  = pwInit + 5 * 8 * 256;              // 8*8*256 dwords
  unsigned* pwRead = pwUpd  + 8 * 8 * 256;              // 4*8*256 dwords

  // pack weights into B-fragment layout (bf16)
  k_pack_w<<<(5 * 8 * 256 + 255) / 256, 256, 0, stream>>>(Winit, pwInit, K_INIT, 5, 8);
  k_pack_w<<<(8 * 8 * 256 + 255) / 256, 256, 0, stream>>>(Wupd,  pwUpd,  2 * H_DIM, 8, 8);
  k_pack_w<<<(4 * 8 * 256 + 255) / 256, 256, 0, stream>>>(Wread, pwRead, H_DIM, 4, 8);

  // edge init (64 edges per block)
  k_edge_init<<<E / 64, 256, 0, stream>>>(atom, bond, srcIdx, pwInit, binit, hA);

  // message passing
  const int segBlocks = (int)(((long long)E * 64 + 255) / 256);
  bf16_t* cur = hA;
  bf16_t* nxt = hB;
  for (int d = 0; d < DEPTH_I; ++d) {
    hipMemsetAsync(sumIn, 0, szS, stream);
    k_segsum<<<segBlocks, 256, 0, stream>>>(cur, dstIdx, sumIn, E);
    k_update<<<E / 64, 256, 0, stream>>>(cur, nxt, sumIn, srcIdx, pwUpd, bupd);
    bf16_t* t = cur; cur = nxt; nxt = t;
  }

  // readout
  hipMemsetAsync(sumIn, 0, szS, stream);
  k_segsum<<<segBlocks, 256, 0, stream>>>(cur, dstIdx, sumIn, E);
  hipMemsetAsync(d_out, 0, (size_t)out_size * sizeof(float), stream);
  k_readout<<<320, 256, 0, stream>>>(sumIn, pwRead, bread, (float*)d_out, N);
}
